// LeapLSTM_20547123544822
// MI455X (gfx1250) — matrix operand order repaired and checked by reading, not compiled, not run on hardware
//
#include <hip/hip_runtime.h>
#include <hip/hip_bf16.h>
#include <math.h>

// ---------------------------------------------------------------- types
typedef __attribute__((ext_vector_type(16))) __bf16 v16bf;
typedef __attribute__((ext_vector_type(8)))  __bf16 v8bf;
typedef __attribute__((ext_vector_type(8)))  float  v8f;
typedef __attribute__((ext_vector_type(4)))  unsigned v4u;
typedef __attribute__((ext_vector_type(8)))  int      v8i;
typedef __attribute__((ext_vector_type(4)))  int      v4i;

union FragBF { v16bf v; v8bf h[2]; };

#if defined(__has_builtin)
#if __has_builtin(__builtin_amdgcn_tensor_load_to_lds) && __has_builtin(__builtin_amdgcn_s_wait_tensorcnt)
#define HAVE_TDM 1
#endif
#endif
#ifndef HAVE_TDM
#define HAVE_TDM 0
#endif

__device__ __forceinline__ v8f vzero8() { v8f z; for (int i = 0; i < 8; ++i) z[i] = 0.f; return z; }

__device__ __forceinline__ v8f wmma_bf(v16bf a, v16bf b, v8f c) {
  // D(16x16 f32) = A(16x32 bf16) * B(32x16 bf16) + C
  return __builtin_amdgcn_wmma_f32_16x16x32_bf16(false, a, false, b, (short)0, c, false, false);
}

__device__ __forceinline__ float sigf(float x) { return 1.f / (1.f + expf(-x)); }

// ---------------------------------------------------------------- problem dims
#define BB   128
#define SS   256
#define DD   512
#define NG   2560   // 4*D (cell gates) + D (leap hidden)
#define NBLK 40     // persistent scan workgroups (2560/64)

// ---------------------------------------------------------------- ws layout (bytes)
static constexpr size_t OFF_BAR  = 0;
static constexpr size_t OFF_X    = 256;                                    // f32 [B,S,D]
static constexpr size_t OFF_XB   = OFF_X    + (size_t)BB*SS*DD*4;          // bf16 [B,S,D]
static constexpr size_t OFF_WCAT = OFF_XB   + (size_t)BB*SS*DD*2;          // bf16 [2560,512]
static constexpr size_t OFF_WREC = OFF_WCAT + (size_t)NG*DD*2;             // bf16 [2560,512]
static constexpr size_t OFF_XG   = OFF_WREC + (size_t)NG*DD*2;             // f32 [B*S,40]
static constexpr size_t OFF_BACK = OFF_XG   + (size_t)BB*SS*40*4;          // f32 [B,S,10]
static constexpr size_t OFF_CNN  = OFF_BACK + (size_t)BB*SS*10*4;          // f32 [B,S,20]
static constexpr size_t OFF_U    = OFF_CNN  + (size_t)BB*SS*20*4;          // f32 [S,B,2]
static constexpr size_t OFF_PRE  = OFF_U    + (size_t)SS*BB*2*4;           // f32 [S,B,2560]
static constexpr size_t OFF_HX   = OFF_PRE  + (size_t)SS*BB*NG*4;          // f32 [B,D]
static constexpr size_t OFF_HXB  = OFF_HX   + (size_t)BB*DD*4;             // bf16 [B,D]
static constexpr size_t OFF_CX   = OFF_HXB  + (size_t)BB*DD*2;             // f32 [B,D]
static constexpr size_t OFF_PST  = OFF_CX   + (size_t)BB*DD*4;             // f32 [B,2560]
static constexpr size_t OFF_SKIP = OFF_PST  + (size_t)BB*NG*4;             // f32 [B,2]
static constexpr size_t OFF_FIN  = OFF_SKIP + (size_t)BB*2*4;              // f32 [B,D]

// ---------------------------------------------------------------- grid barrier
__device__ __forceinline__ void grid_barrier(unsigned* cnt, unsigned* gen, unsigned nb) {
  __syncthreads();
  if (threadIdx.x == 0) {
    __threadfence();
    unsigned g = __hip_atomic_load(gen, __ATOMIC_RELAXED, __HIP_MEMORY_SCOPE_AGENT);
    unsigned arrived = __hip_atomic_fetch_add(cnt, 1u, __ATOMIC_ACQ_REL, __HIP_MEMORY_SCOPE_AGENT);
    if (arrived == nb - 1u) {
      __hip_atomic_store(cnt, 0u, __ATOMIC_RELAXED, __HIP_MEMORY_SCOPE_AGENT);
      __hip_atomic_store(gen, g + 1u, __ATOMIC_RELEASE, __HIP_MEMORY_SCOPE_AGENT);
    } else {
      while (__hip_atomic_load(gen, __ATOMIC_ACQUIRE, __HIP_MEMORY_SCOPE_AGENT) == g) {
        __builtin_amdgcn_s_sleep(2);
      }
    }
    __threadfence();
  }
  __syncthreads();
}

// ---------------------------------------------------------------- kernels
__global__ void k_init(unsigned* bar) {
  if (threadIdx.x < 8) bar[threadIdx.x] = 0u;
}

// build bf16 weight matrices: Wcat[r][k] (xt path) and Wrec[r][k] (hx path)
__global__ void k_convert(const float* __restrict__ cell_Wih, const float* __restrict__ cell_Whh,
                          const float* __restrict__ leap_W1, __bf16* __restrict__ Wcat,
                          __bf16* __restrict__ Wrec) {
  int i = blockIdx.x * 256 + threadIdx.x;
  if (i >= NG * DD) return;
  int r = i >> 9, k = i & 511;
  float vc, vr;
  if (r < 2048) { vc = cell_Wih[i]; vr = cell_Whh[i]; }
  else {
    int d = r - 2048;
    vc = leap_W1[(size_t)d * 1054 + 512 + k];   // xt slice of W1
    vr = leap_W1[(size_t)d * 1054 + k];         // hx slice of W1
  }
  Wcat[i] = (__bf16)vc;
  Wrec[i] = (__bf16)vr;
}

// embedding gather + sum over V=16 codes; write f32 and bf16 copies
__global__ void k_embed(const int* __restrict__ seqs, const float* __restrict__ emb,
                        float* __restrict__ x, __bf16* __restrict__ xb) {
  int bs = blockIdx.x;                 // b*S + s
  int d0 = threadIdx.x * 4;            // 128 threads * 4
  const int* code = seqs + (size_t)bs * 16;
  float a0 = 0.f, a1 = 0.f, a2 = 0.f, a3 = 0.f;
  for (int v = 0; v < 16; ++v) {
    const float* row = emb + (size_t)code[v] * DD + d0;
    a0 += row[0]; a1 += row[1]; a2 += row[2]; a3 += row[3];
  }
  size_t base = (size_t)bs * DD + d0;
  x[base] = a0; x[base + 1] = a1; x[base + 2] = a2; x[base + 3] = a3;
  xb[base] = (__bf16)a0; xb[base + 1] = (__bf16)a1;
  xb[base + 2] = (__bf16)a2; xb[base + 3] = (__bf16)a3;
}

// reverse-LSTM input projection: xg[m][j] = x[m,:] . rev_Wih[j,:]
__global__ void k_revproj(const float* __restrict__ x, const float* __restrict__ Wih,
                          float* __restrict__ xg) {
  int i = blockIdx.x * 256 + threadIdx.x;
  if (i >= BB * SS * 40) return;
  int j = i % 40, m = i / 40;
  const float* xr = x + (size_t)m * DD;
  const float* w = Wih + (size_t)j * DD;
  float a = 0.f;
  for (int d = 0; d < DD; ++d) a += xr[d] * w[d];
  xg[i] = a;
}

// reverse LSTM scan (hidden=10), processes t = S-1 .. 0
__global__ void k_revscan(const float* __restrict__ xg, const float* __restrict__ Whh,
                          float* __restrict__ back) {
  __shared__ float hbuf[32 * 10];
  int tid = threadIdx.x;                    // 320 threads: 32 b * 10 h
  int bl = tid / 10, h = tid % 10;
  int b = blockIdx.x * 32 + bl;
  hbuf[tid] = 0.f;
  float c = 0.f;
  __syncthreads();
  for (int t = SS - 1; t >= 0; --t) {
    const float* xr = xg + ((size_t)b * SS + t) * 40;
    float g[4];
    for (int q = 0; q < 4; ++q) {
      int j = q * 10 + h;
      float a = xr[j];
      const float* w = Whh + j * 10;
      for (int p = 0; p < 10; ++p) a += hbuf[bl * 10 + p] * w[p];
      g[q] = a;
    }
    c = sigf(g[1]) * c + sigf(g[0]) * tanhf(g[2]);
    float hn = sigf(g[3]) * tanhf(c);
    __syncthreads();
    hbuf[bl * 10 + h] = hn;
    back[((size_t)b * SS + t) * 10 + h] = hn;
    __syncthreads();
  }
}

// Conv1d D->20, k=3, pad=1, ReLU
__global__ void k_conv(const float* __restrict__ x, const float* __restrict__ w,
                       float* __restrict__ cnn) {
  int i = blockIdx.x * 256 + threadIdx.x;
  if (i >= BB * SS * 20) return;
  int oc = i % 20, t = (i / 20) % SS, b = i / (20 * SS);
  float a = 0.f;
  for (int k = 0; k < 3; ++k) {
    int tt = t + k - 1;
    if (tt < 0 || tt >= SS) continue;
    const float* xr = x + ((size_t)b * SS + tt) * DD;
    const float* wr = w + (size_t)oc * (DD * 3) + k;
    for (int d = 0; d < DD; ++d) a += xr[d] * wr[d * 3];
  }
  cnn[i] = a > 0.f ? a : 0.f;
}

// threefry2x32-20, key(42) = (0,42); jax.random.uniform bit trick
__device__ __forceinline__ unsigned rotl32(unsigned x, int r) { return (x << r) | (x >> (32 - r)); }
__device__ __forceinline__ float bits01(unsigned b) {
  return __uint_as_float((b >> 9) | 0x3F800000u) - 1.0f;
}
__global__ void k_gumbel(float* __restrict__ U) {
  unsigned i = blockIdx.x * 256 + threadIdx.x;
  if (i >= 32768u) return;
  const unsigned ks0 = 0u, ks1 = 42u;
  const unsigned ks2 = 0x1BD11BDAu ^ ks0 ^ ks1;
  unsigned x0 = i + ks0, x1 = (32768u + i) + ks1;
  const int r1[4] = {13, 15, 26, 6};
  const int r2[4] = {17, 29, 16, 24};
  for (int j = 0; j < 4; ++j) { x0 += x1; x1 = rotl32(x1, r1[j]); x1 ^= x0; }
  x0 += ks1; x1 += ks2 + 1u;
  for (int j = 0; j < 4; ++j) { x0 += x1; x1 = rotl32(x1, r2[j]); x1 ^= x0; }
  x0 += ks2; x1 += ks0 + 2u;
  for (int j = 0; j < 4; ++j) { x0 += x1; x1 = rotl32(x1, r1[j]); x1 ^= x0; }
  x0 += ks0; x1 += ks1 + 3u;
  for (int j = 0; j < 4; ++j) { x0 += x1; x1 = rotl32(x1, r2[j]); x1 ^= x0; }
  x0 += ks1; x1 += ks2 + 4u;
  for (int j = 0; j < 4; ++j) { x0 += x1; x1 = rotl32(x1, r1[j]); x1 ^= x0; }
  x0 += ks2; x1 += ks0 + 5u;
  U[i] = bits01(x0);
  U[i + 32768u] = bits01(x1);
}

// time-parallel GEMM: Pre[s][b][n] = X[b*S+s,:] @ Wcat[n,:]  (bf16 WMMA, f32 acc)
__global__ void __launch_bounds__(256) k_gemm_pre(const __bf16* __restrict__ X,
                                                  const __bf16* __restrict__ Wcat,
                                                  float* __restrict__ Pre) {
  const int mb = blockIdx.x * 64;
  const int nb = blockIdx.y * 64;
  const int lane = threadIdx.x & 31, wave = threadIdx.x >> 5;
  const int half = lane >> 4, lrow = lane & 15;
  const int mt = mb + (wave & 3) * 16;
  const int nt = nb + (wave >> 2) * 32;
  v8f acc0 = vzero8(), acc1 = vzero8();
  for (int kk = 0; kk < DD; kk += 32) {
    FragBF a, b0, b1;
    const __bf16* ap = X + (size_t)(mt + lrow) * DD + kk + half * 16;
    a.h[0] = *(const v8bf*)ap;
    a.h[1] = *(const v8bf*)(ap + 8);
    const __bf16* bp0 = Wcat + (size_t)(nt + lrow) * DD + kk + half * 16;
    b0.h[0] = *(const v8bf*)bp0;
    b0.h[1] = *(const v8bf*)(bp0 + 8);
    const __bf16* bp1 = Wcat + (size_t)(nt + 16 + lrow) * DD + kk + half * 16;
    b1.h[0] = *(const v8bf*)bp1;
    b1.h[1] = *(const v8bf*)(bp1 + 8);
    acc0 = wmma_bf(a.v, b0.v, acc0);
    acc1 = wmma_bf(a.v, b1.v, acc1);
  }
  for (int r = 0; r < 8; ++r) {
    int m = mt + r + 8 * half;
    int b = m >> 8, s = m & 255;
    size_t rowbase = ((size_t)s * BB + b) * NG;
    Pre[rowbase + nt + lrow] = acc0[r];
    Pre[rowbase + nt + 16 + lrow] = acc1[r];
  }
}

// add small leap contributions: back@W1_b.T + cnn@W1_c.T + b1
__global__ void k_addsmall(const float* __restrict__ back, const float* __restrict__ cnn,
                           const float* __restrict__ W1, const float* __restrict__ b1,
                           float* __restrict__ Pre) {
  size_t i = (size_t)blockIdx.x * 256 + threadIdx.x;   // over S*B*512
  int d = (int)(i & 511);
  int sb = (int)(i >> 9);          // s*128 + b
  int b = sb & 127, s = sb >> 7;
  const float* bk = back + ((size_t)b * SS + s) * 10;
  const float* cn = cnn + ((size_t)b * SS + s) * 20;
  const float* w = W1 + (size_t)d * 1054;
  float a = b1[d];
  for (int j = 0; j < 10; ++j) a += bk[j] * w[1024 + j];
  for (int j = 0; j < 20; ++j) a += cn[j] * w[1034 + j];
  Pre[(size_t)sb * NG + 2048 + d] += a;
}

// persistent sequential scan: 40 WGs, grid barriers, WMMA recurrent GEMM from LDS weights
__global__ void __launch_bounds__(256) k_scan(
    const __bf16* __restrict__ Wrec, const float* __restrict__ Pre,
    const float* __restrict__ W2, const float* __restrict__ b2,
    const float* __restrict__ U, const int* __restrict__ lengths,
    float* __restrict__ hx, __bf16* __restrict__ hxb, float* __restrict__ cx,
    float* __restrict__ pst, float* __restrict__ skip, float* __restrict__ fin,
    unsigned* barcnt, unsigned* bargen) {
  extern __shared__ char smem[];
  __bf16* Bs = (__bf16*)smem;                       // 64 rows x 512 k  (64 KB)
  float* red = (float*)(smem + 64 * DD * 2);        // 512 floats reduction buffer

  const int tid = threadIdx.x;
  const int blk = blockIdx.x;
  const int nb = blk * 64;

  // ---- stage this WG's Wrec tile into LDS once (constant across all 256 steps)
#if HAVE_TDM
  if (tid < 32) {
    // Tensor DMA descriptor (CDNA5 ISA):
    // 2D tile: tile_dim0=512 elems (one K row), tile_dim1=64 rows, data_size=2B,
    // tensor_dim0=512, tensor_dim1=64, tensor_dim0_stride=512 -> contiguous 64KB.
    unsigned lds_addr = (unsigned)(unsigned long long)
        (__attribute__((address_space(3))) char*)smem;
    unsigned long long ga = (unsigned long long)(const void*)(Wrec + (size_t)nb * DD);
    v4u g0;
    g0[0] = 1u;                                               // count=1, user mode
    g0[1] = lds_addr;                                         // lds_addr [63:32]
    g0[2] = (unsigned)(ga & 0xFFFFFFFFull);                   // global_addr lo
    g0[3] = (unsigned)((ga >> 32) & 0x1FFFFFFull) | (2u << 30); // ga hi, type=2
    v8i g1;
    g1[0] = 0x10000;               // workgroup_mask=0, data_size=1 (2 bytes)
    g1[1] = (int)(512u << 16);     // tensor_dim0 lo16 in [31:16]
    g1[2] = (int)(64u << 16);      // tensor_dim0 hi16=0 | tensor_dim1 lo16=64
    g1[3] = (int)(512u << 16);     // tensor_dim1 hi16=0 | tile_dim0=512
    g1[4] = 64;                    // tile_dim1=64 | tile_dim2=0
    g1[5] = 512;                   // tensor_dim0_stride lo32
    g1[6] = 0;                     // stride hi | tensor_dim1_stride lo
    g1[7] = 0;
    v4i gz4 = {0, 0, 0, 0};
    v8i gz8 = {0, 0, 0, 0, 0, 0, 0, 0};
    __builtin_amdgcn_tensor_load_to_lds(g0, g1, gz4, gz4, gz8, 0);
    __builtin_amdgcn_s_wait_tensorcnt(0);
  }
#else
  for (int i = tid; i < 64 * DD / 8; i += 256) {
    *(v8bf*)(Bs + (size_t)i * 8) = *(const v8bf*)(Wrec + (size_t)nb * DD + (size_t)i * 8);
  }
#endif
  // zero-init recurrent state
  for (int i = blk * 256 + tid; i < BB * DD; i += NBLK * 256) {
    hx[i] = 0.f; cx[i] = 0.f; hxb[i] = (__bf16)0.f;
  }
  __syncthreads();
  grid_barrier(barcnt, bargen, NBLK);

  const int lane = tid & 31;
  const int wave = tid >> 5;
  const int half = lane >> 4;
  const int lrow = lane & 15;
  const int m0 = wave * 16;

  for (int t = 0; t < SS; ++t) {
    // software prefetch of next step's Pre tile (HBM-streamed, 256B per row)
    if (t + 1 < SS) {
      const float* np = Pre + ((size_t)(t + 1) * BB + (tid >> 1)) * NG + nb + (tid & 1) * 32;
      __builtin_prefetch(np, 0, 1);
    }
    // ---- phase 1: pst[m, nb..nb+63] = hx @ Wrec_tile.T + Pre[t]
    v8f acc[4];
    for (int s2 = 0; s2 < 4; ++s2) acc[s2] = vzero8();
    for (int kk = 0; kk < DD; kk += 32) {
      FragBF a;
      const __bf16* ap = hxb + (size_t)(m0 + lrow) * DD + kk + half * 16;
      a.h[0] = *(const v8bf*)ap;
      a.h[1] = *(const v8bf*)(ap + 8);
      for (int s2 = 0; s2 < 4; ++s2) {
        FragBF b;
        const __bf16* bp = Bs + (size_t)(s2 * 16 + lrow) * DD + kk + half * 16;
        b.h[0] = *(const v8bf*)bp;
        b.h[1] = *(const v8bf*)(bp + 8);
        acc[s2] = wmma_bf(a.v, b.v, acc[s2]);
      }
    }
    const float* pt = Pre + (size_t)t * BB * NG;
    for (int s2 = 0; s2 < 4; ++s2) {
      for (int r = 0; r < 8; ++r) {
        int m = m0 + r + 8 * half;
        int n = nb + s2 * 16 + lrow;
        pst[(size_t)m * NG + n] = acc[s2][r] + pt[(size_t)m * NG + n];
      }
    }
    grid_barrier(barcnt, bargen, NBLK);

    // ---- phase 2: gumbel-softmax skip gate (first 32 WGs, 4 batch rows each)
    if (blk < 32) {
      for (int q = 0; q < 4; ++q) {
        int b = blk * 4 + q;
        float p0 = 0.f, p1 = 0.f;
        const float* lp = pst + (size_t)b * NG + 2048;
        for (int d = tid; d < DD; d += 256) {
          float v = lp[d];
          v = v > 0.f ? v : 0.f;                  // ReLU
          p0 += v * W2[d];
          p1 += v * W2[DD + d];
        }
        red[tid] = p0; red[256 + tid] = p1;
        __syncthreads();
        for (int st = 128; st > 0; st >>= 1) {
          if (tid < st) { red[tid] += red[tid + st]; red[256 + tid] += red[256 + tid + st]; }
          __syncthreads();
        }
        if (tid == 0) {
          float l0 = red[0] + b2[0], l1 = red[256] + b2[1];
          float mx = fmaxf(l0, l1);
          float lse = mx + logf(expf(l0 - mx) + expf(l1 - mx));
          float u0 = U[((size_t)t * BB + b) * 2 + 0];
          float u1 = U[((size_t)t * BB + b) * 2 + 1];
          float y0 = (l0 - lse) - logf(-logf(u0 + 1e-20f) + 1e-20f);
          float y1 = (l1 - lse) - logf(-logf(u1 + 1e-20f) + 1e-20f);
          float gm = fmaxf(y0, y1);
          float e0 = expf((y0 - gm) * 1e5f);
          float e1 = expf((y1 - gm) * 1e5f);
          float inv = 1.f / (e0 + e1);
          skip[2 * b] = e0 * inv;
          skip[2 * b + 1] = e1 * inv;
        }
        __syncthreads();
      }
    }
    grid_barrier(barcnt, bargen, NBLK);

    // ---- phase 3: LSTM cell + skip blend + length gather
    for (int i = blk * 256 + tid; i < BB * DD; i += NBLK * 256) {
      int b = i >> 9, d = i & 511;
      const float* pb = pst + (size_t)b * NG;
      float ig = pb[d], fg = pb[512 + d], gg = pb[1024 + d], og = pb[1536 + d];
      float cnew = sigf(fg) * cx[i] + sigf(ig) * tanhf(gg);
      float hstep = sigf(og) * tanhf(cnew);
      float hnew = hx[i] * skip[2 * b + 1] + hstep * skip[2 * b];
      cx[i] = cnew;
      hx[i] = hnew;
      hxb[i] = (__bf16)hnew;
      if (t == lengths[b] - 1) fin[i] = hnew;
    }
    grid_barrier(barcnt, bargen, NBLK);
  }
}

// final linear: out[b][j] = fin[b,:] . out_W[j,:] + out_b[j]
__global__ void k_out(const float* __restrict__ fin, const float* __restrict__ out_W,
                      const float* __restrict__ out_b, float* __restrict__ out) {
  int p = threadIdx.x;              // 256 = 128 b * 2 j
  int b = p >> 1, j = p & 1;
  const float* f = fin + (size_t)b * DD;
  const float* w = out_W + (size_t)j * DD;
  float a = out_b[j];
  for (int d = 0; d < DD; ++d) a += f[d] * w[d];
  out[b * 2 + j] = a;
}

// ---------------------------------------------------------------- launch
extern "C" void kernel_launch(void* const* d_in, const int* in_sizes, int n_in,
                              void* d_out, int out_size, void* d_ws, size_t ws_size,
                              hipStream_t stream) {
  (void)in_sizes; (void)n_in; (void)out_size; (void)ws_size;

  const int*   seqs     = (const int*)d_in[0];
  const int*   lengths  = (const int*)d_in[2];
  const float* emb      = (const float*)d_in[5];
  const float* rev_Wih  = (const float*)d_in[6];
  const float* rev_Whh  = (const float*)d_in[7];
  const float* conv_w   = (const float*)d_in[8];
  const float* leap_W1  = (const float*)d_in[9];
  const float* leap_b1  = (const float*)d_in[10];
  const float* leap_W2  = (const float*)d_in[11];
  const float* leap_b2  = (const float*)d_in[12];
  const float* cell_Wih = (const float*)d_in[13];
  const float* cell_Whh = (const float*)d_in[14];
  const float* out_W    = (const float*)d_in[15];
  const float* out_b    = (const float*)d_in[16];

  char* ws = (char*)d_ws;
  unsigned* bar  = (unsigned*)(ws + OFF_BAR);
  float*    x    = (float*)(ws + OFF_X);
  __bf16*   xb   = (__bf16*)(ws + OFF_XB);
  __bf16*   Wcat = (__bf16*)(ws + OFF_WCAT);
  __bf16*   Wrec = (__bf16*)(ws + OFF_WREC);
  float*    xg   = (float*)(ws + OFF_XG);
  float*    back = (float*)(ws + OFF_BACK);
  float*    cnn  = (float*)(ws + OFF_CNN);
  float*    U    = (float*)(ws + OFF_U);
  float*    Pre  = (float*)(ws + OFF_PRE);
  float*    hx   = (float*)(ws + OFF_HX);
  __bf16*   hxb  = (__bf16*)(ws + OFF_HXB);
  float*    cx   = (float*)(ws + OFF_CX);
  float*    pst  = (float*)(ws + OFF_PST);
  float*    skp  = (float*)(ws + OFF_SKIP);
  float*    fin  = (float*)(ws + OFF_FIN);

  k_init<<<1, 64, 0, stream>>>(bar);
  k_convert<<<(NG * DD + 255) / 256, 256, 0, stream>>>(cell_Wih, cell_Whh, leap_W1, Wcat, Wrec);
  k_embed<<<BB * SS, 128, 0, stream>>>(seqs, emb, x, xb);
  k_revproj<<<(BB * SS * 40 + 255) / 256, 256, 0, stream>>>(x, rev_Wih, xg);
  k_revscan<<<4, 320, 0, stream>>>(xg, rev_Whh, back);
  k_conv<<<(BB * SS * 20 + 255) / 256, 256, 0, stream>>>(x, conv_w, cnn);
  k_gumbel<<<128, 256, 0, stream>>>(U);
  {
    dim3 g(BB * SS / 64, NG / 64);   // 512 x 40
    k_gemm_pre<<<g, 256, 0, stream>>>(xb, Wcat, Pre);
  }
  k_addsmall<<<(SS * BB * DD) / 256, 256, 0, stream>>>(back, cnn, leap_W1, leap_b1, Pre);
  k_scan<<<NBLK, 256, 64 * DD * 2 + 2048, stream>>>(Wrec, Pre, leap_W2, leap_b2, U, lengths,
                                                    hx, hxb, cx, pst, skp, fin, bar, bar + 1);
  k_out<<<1, 256, 0, stream>>>(fin, out_W, out_b, (float*)d_out);
}
